// SelfAttention_65481071403879
// MI455X (gfx1250) — compile-verified, matrix operand order checked by reading, not run on hardware
//
#include <hip/hip_runtime.h>
#include <hip/hip_bf16.h>

typedef __bf16 bf16;
typedef __attribute__((ext_vector_type(16))) __bf16 v16bf;
typedef __attribute__((ext_vector_type(8)))  float  v8f;

struct __align__(16) U128 { unsigned int x, y, z, w; };

// Load one 16x32 bf16 WMMA operand fragment (8 VGPRs) for this lane.
// p already points at &Mtx[row_of_this_lane][k + (lane>>4)*8].
// VGPR0..3 <- K=base..base+7 (contiguous 16B), VGPR4..7 <- K=base+16..base+23.
__device__ __forceinline__ v16bf ld_frag(const bf16* p) {
    union { v16bf v; U128 q[2]; } u;
    u.q[0] = *(const U128*)(p);
    u.q[1] = *(const U128*)(p + 16);
    return u.v;
}

#define WMMA_BF16(a, b, c) \
    __builtin_amdgcn_wmma_f32_16x16x32_bf16(false, (a), false, (b), (short)0, (c), false, false)

// C[M,N] = A[M,K] @ B[N,K]^T   (both row-major; "NT" layout)
// Block = 4 waves (128 thr) covering 128x128; wave = 64x64 = 4x4 WMMA tiles
// (16 WMMAs per 8 fragment loads -> 16B/lane of cache traffic per WMMA).
// OUTM: 0 = fp32 out, 1 = bf16 out, 2 = bf16 out stored transposed per batch of 2048 rows
// BIAS: add bias[col]; CAUSAL: scale then mask col>row with -1e30 (fp32 out only)
// TRIA: A rows are causally lower-triangular -> clamp K at (blockIdx.y+1)*128
template<int OUTM, bool BIAS, bool CAUSAL, bool TRIA>
__global__ __launch_bounds__(128)
void gemm_nt_wmma(const bf16* __restrict__ A, const bf16* __restrict__ B,
                  void* __restrict__ C, const float* __restrict__ bias,
                  int K, int lda, int ldb, int ldc,
                  long long sA, long long sB, long long sC, float scale) {
    const int bz   = blockIdx.z;
    const bf16* Ab = A + (size_t)bz * sA;
    const bf16* Bb = B + (size_t)bz * sB;

    const int lane = threadIdx.x & 31;
    const int w    = threadIdx.x >> 5;     // 4 waves: 2 along M x 2 along N
    const int lr   = lane & 15;            // row/col within 16-tile
    const int lh   = lane >> 4;            // lane-half selects K-octet / M-half
    const int m0   = blockIdx.y * 128 + (w & 1) * 64;
    const int n0   = blockIdx.x * 128 + (w >> 1) * 64;

    if (CAUSAL) {
        // Fully masked block: skip the whole K loop, just fill -1e30.
        const int mTop  = blockIdx.y * 128;
        const int nLeft = blockIdx.x * 128;
        if (nLeft > mTop + 127) {
            float* Cf = (float*)C + (size_t)bz * sC;
            const float4 fill = {-1e30f, -1e30f, -1e30f, -1e30f};
            for (int t = threadIdx.x; t < 128 * 32; t += 128) {  // 128 rows x 32 float4
                float4* rp = (float4*)(Cf + (size_t)(mTop + (t >> 5)) * ldc + nLeft);
                rp[t & 31] = fill;
            }
            return;
        }
    }

    int kmax = K;
    if (TRIA) { const int lim = (blockIdx.y + 1) * 128; kmax = lim < K ? lim : K; }

    const bf16* arow[4];
    const bf16* brow[4];
    #pragma unroll
    for (int i = 0; i < 4; ++i) {
        arow[i] = Ab + (size_t)(m0 + 16 * i + lr) * lda + lh * 8;
        brow[i] = Bb + (size_t)(n0 + 16 * i + lr) * ldb + lh * 8;
    }

    const v8f z = {0.f,0.f,0.f,0.f,0.f,0.f,0.f,0.f};
    v8f acc[4][4];
    #pragma unroll
    for (int i = 0; i < 4; ++i)
        #pragma unroll
        for (int j = 0; j < 4; ++j) acc[i][j] = z;

    #pragma unroll 1
    for (int k = 0; k < kmax; k += 32) {
        v16bf af[4], bfrag[4];
        #pragma unroll
        for (int i = 0; i < 4; ++i) af[i] = ld_frag(arow[i] + k);
        #pragma unroll
        for (int j = 0; j < 4; ++j) bfrag[j] = ld_frag(brow[j] + k);
        #pragma unroll
        for (int i = 0; i < 4; ++i)
            #pragma unroll
            for (int j = 0; j < 4; ++j)
                acc[i][j] = WMMA_BF16(af[i], bfrag[j], acc[i][j]);
    }

    float* Cf = (float*)C + (size_t)bz * sC;
    bf16*  Cb = (OUTM == 1) ? (bf16*)C + (size_t)bz * sC : (bf16*)C;

    #pragma unroll
    for (int i = 0; i < 4; ++i) {
        #pragma unroll
        for (int j = 0; j < 4; ++j) {
            const int col = n0 + 16 * j + lr;
            const float badd = BIAS ? bias[col] : 0.0f;
            #pragma unroll
            for (int e = 0; e < 8; ++e) {
                const int row = m0 + 16 * i + lh * 8 + e;  // f32 C/D layout: VGPR e, lane-half
                float v = acc[i][j][e] + badd;
                if (CAUSAL) { v *= scale; if (col > row) v = -1e30f; }
                if (OUTM == 0) {
                    Cf[(size_t)row * ldc + col] = v;
                } else if (OUTM == 1) {
                    Cb[(size_t)row * ldc + col] = (bf16)v;
                } else { // transposed per-batch store: row = b*2048 + s  ->  Ct[b][col][s]
                    Cb[(size_t)(row >> 11) * sC + (size_t)col * ldc + (row & 2047)] = (bf16)v;
                }
            }
        }
    }
}

__global__ void cvt_f32_to_bf16(const float* __restrict__ in, bf16* __restrict__ out, int n4) {
    const int i = blockIdx.x * blockDim.x + threadIdx.x;
    if (i < n4) {
        const float4 f = ((const float4*)in)[i];
        bf16* o = out + (size_t)i * 4;
        o[0] = (bf16)f.x; o[1] = (bf16)f.y; o[2] = (bf16)f.z; o[3] = (bf16)f.w;
    }
}

// One block per row; row length 2048, 256 threads x 8 elements.
// Reads fp32 scores, writes normalized probs as bf16 IN PLACE over the same row
// (all reads complete before any write; bf16 row occupies the first half of the fp32 row).
__global__ __launch_bounds__(256)
void softmax_inplace_bf16(float* __restrict__ scores, int S) {
    __shared__ float red[256];
    const int row = blockIdx.x;
    float* src = scores + (size_t)row * S;
    bf16*  dst = (bf16*)src;
    const int t = threadIdx.x;

    float v[8];
    float mx = -3.4e38f;
    #pragma unroll
    for (int i = 0; i < 8; ++i) { v[i] = src[t + (i << 8)]; mx = fmaxf(mx, v[i]); }
    red[t] = mx; __syncthreads();
    for (int s = 128; s > 0; s >>= 1) { if (t < s) red[t] = fmaxf(red[t], red[t + s]); __syncthreads(); }
    mx = red[0]; __syncthreads();

    float sum = 0.f;
    #pragma unroll
    for (int i = 0; i < 8; ++i) { v[i] = __expf(v[i] - mx); sum += v[i]; }
    red[t] = sum; __syncthreads();
    for (int s = 128; s > 0; s >>= 1) { if (t < s) red[t] += red[t + s]; __syncthreads(); }
    const float inv = 1.0f / red[0];
    __syncthreads();

    #pragma unroll
    for (int i = 0; i < 8; ++i) dst[t + (i << 8)] = (bf16)(v[i] * inv);
}

extern "C" void kernel_launch(void* const* d_in, const int* in_sizes, int n_in,
                              void* d_out, int out_size, void* d_ws, size_t ws_size,
                              hipStream_t stream) {
    (void)in_sizes; (void)n_in; (void)out_size; (void)ws_size;
    constexpr int D = 1024, Bt = 4, S = 2048;
    constexpr int MS = Bt * S;                      // 8192 flattened rows

    const float* X  = (const float*)d_in[0];
    const float* Rr = (const float*)d_in[1];
    const float* Ee = (const float*)d_in[2];
    const float* Wq = (const float*)d_in[3];
    const float* bq = (const float*)d_in[4];
    const float* Wk = (const float*)d_in[5];
    const float* bk = (const float*)d_in[6];
    const float* Wv = (const float*)d_in[7];
    const float* bv = (const float*)d_in[8];
    // d_in[9] = bool causal mask: implied analytically, not read.

    char* p = (char*)d_ws;
    auto take = [&](size_t bytes) { char* r = p; p += (bytes + 255) & ~(size_t)255; return r; };
    bf16*  Xb  = (bf16*)take((size_t)MS * D * 2);
    bf16*  Wqb = (bf16*)take((size_t)D * D * 2);
    bf16*  Wkb = (bf16*)take((size_t)D * D * 2);
    bf16*  Wvb = (bf16*)take((size_t)D * D * 2);
    bf16*  Rb  = (bf16*)take((size_t)D * D * 2);
    bf16*  Eb  = (bf16*)take((size_t)D * D * 2);
    bf16*  T   = (bf16*)take((size_t)MS * D * 2);  // reused for q-temp then k-temp
    bf16*  Qb  = (bf16*)take((size_t)MS * D * 2);
    bf16*  Kb  = (bf16*)take((size_t)MS * D * 2);
    bf16*  Vtb = (bf16*)take((size_t)MS * D * 2);  // V transposed: [b][d][s]
    float* Sc  = (float*)take((size_t)Bt * S * S * 4);

    const dim3 cblk(256);
    const dim3 gblk(128);
    const int nW4 = D * D / 4;
    cvt_f32_to_bf16<<<dim3((MS * D / 4 + 255) / 256), cblk, 0, stream>>>(X,  Xb,  MS * D / 4);
    cvt_f32_to_bf16<<<dim3((nW4 + 255) / 256),        cblk, 0, stream>>>(Wq, Wqb, nW4);
    cvt_f32_to_bf16<<<dim3((nW4 + 255) / 256),        cblk, 0, stream>>>(Wk, Wkb, nW4);
    cvt_f32_to_bf16<<<dim3((nW4 + 255) / 256),        cblk, 0, stream>>>(Wv, Wvb, nW4);
    cvt_f32_to_bf16<<<dim3((nW4 + 255) / 256),        cblk, 0, stream>>>(Rr, Rb,  nW4);
    cvt_f32_to_bf16<<<dim3((nW4 + 255) / 256),        cblk, 0, stream>>>(Ee, Eb,  nW4);

    const dim3 gProj(D / 128, MS / 128, 1);
    // T = X@Wq^T + bq ; Q = T@R^T
    gemm_nt_wmma<1, true,  false, false><<<gProj, gblk, 0, stream>>>(Xb, Wqb, T,  bq,      D, D, D, D, 0, 0, 0, 1.0f);
    gemm_nt_wmma<1, false, false, false><<<gProj, gblk, 0, stream>>>(T,  Rb,  Qb, nullptr, D, D, D, D, 0, 0, 0, 1.0f);
    // T = X@Wk^T + bk ; K = T@E^T
    gemm_nt_wmma<1, true,  false, false><<<gProj, gblk, 0, stream>>>(Xb, Wkb, T,  bk,      D, D, D, D, 0, 0, 0, 1.0f);
    gemm_nt_wmma<1, false, false, false><<<gProj, gblk, 0, stream>>>(T,  Eb,  Kb, nullptr, D, D, D, D, 0, 0, 0, 1.0f);
    // Vt = (X@Wv^T + bv)^T  (per-batch transposed store)
    gemm_nt_wmma<2, true,  false, false><<<gProj, gblk, 0, stream>>>(Xb, Wvb, Vtb, bv, D, D, D, S, 0, 0, (long long)D * S, 1.0f);

    // scores[b] = Q_b @ K_b^T * (1/32), causal masked (fully-masked blocks skip compute)
    gemm_nt_wmma<0, false, true, false><<<dim3(S / 128, S / 128, Bt), gblk, 0, stream>>>(
        Qb, Kb, Sc, nullptr, D, D, D, S,
        (long long)S * D, (long long)S * D, (long long)S * S, 0.03125f);

    // row softmax, probs -> bf16 in place
    softmax_inplace_bf16<<<dim3(Bt * S), cblk, 0, stream>>>(Sc, S);

    // out[b] = P_b @ Vt_b^T  (K clamped to causal extent per block row)
    gemm_nt_wmma<0, false, false, true><<<dim3(D / 128, S / 128, Bt), gblk, 0, stream>>>(
        (const bf16*)Sc, Vtb, d_out, nullptr, S, 2 * S, S, D,
        (long long)S * 2 * S, (long long)D * S, (long long)S * D, 1.0f);
}
